// Tem_Agg_Layer_8366596292955
// MI455X (gfx1250) — compile-verified, matrix operand order checked by reading, not run on hardware
//
#include <hip/hip_runtime.h>
#include <hip/hip_bf16.h>

typedef __attribute__((ext_vector_type(16))) __bf16 v16bf;
typedef __attribute__((ext_vector_type(8)))  float  v8f;
typedef __attribute__((ext_vector_type(4)))  float  f32x4;

#define IN_DIM  256
#define OUT_DIM 128
#define KSTEPS  (IN_DIM / 32)          // 8
#define NTILES  (OUT_DIM / 16)         // 8
#define NFRAG   (KSTEPS * NTILES * 32) // 2048 fragments -> 64 KB LDS

// ---------------------------------------------------------------------------
// Phase 1: z = F @ W^T via v_wmma_f32_16x16x32_bf16.
// One wave computes a 16x128 row block. K = 256 -> 8 k-steps of 32.
//
// WMMA 16-bit fragment layout (ISA 7.12.2), half h = lane/16, l16 = lane%16:
//   A (16x32, M in lanes): elems [0..7]  = F[m0+l16][32ks + 8h + 0..7]
//                          elems [8..15] = F[m0+l16][32ks + 8h + 16..23]
//   B (32x16, N in lanes): mirrored; B[k][n] = W[n][k] -> same two contiguous
//                          float8 runs along a W row.
//
// W (128x256 f32) is converted once per workgroup into LDS as *pre-packed*
// per-lane bf16 fragments, fragment-major: Wfrag[((ks*8+nt)*32 + lane)] is a
// contiguous 32B chunk -> inner loop is 2x ds_load_b128 + 1 wmma per tile.
// All 8 B fragments of a k-step are preloaded into distinct registers so the
// 8 WMMAs issue back-to-back behind a single dscnt wait.
// ---------------------------------------------------------------------------
__global__ void __launch_bounds__(256)
gemm_z_kernel(const float* __restrict__ F, const float* __restrict__ W,
              float* __restrict__ Z, int n_src, int row_blocks) {
  __shared__ v16bf Wfrag[NFRAG];       // 64 KB

  const int tid  = threadIdx.x;
  const int lane = tid & 31;
  const int wave = tid >> 5;

  // ---- cooperative prepack of W into LDS (before any wave exits) ----
#pragma unroll
  for (int i = 0; i < NFRAG / 256; ++i) {
    const int f      = i * 256 + tid;
    const int lane_f = f & 31;
    const int nt     = (f >> 5) & 7;
    const int ks     = f >> 8;
    const int h      = lane_f >> 4;
    const int l16f   = lane_f & 15;
    const int klo    = ks * 32 + h * 8;
    const float* __restrict__ Wr = W + (size_t)(nt * 16 + l16f) * IN_DIM;
    float wv[16];
    *(f32x4*)(wv + 0)  = *(const f32x4*)(Wr + klo);
    *(f32x4*)(wv + 4)  = *(const f32x4*)(Wr + klo + 4);
    *(f32x4*)(wv + 8)  = *(const f32x4*)(Wr + klo + 16);
    *(f32x4*)(wv + 12) = *(const f32x4*)(Wr + klo + 20);
    v16bf frag;
#pragma unroll
    for (int j = 0; j < 16; ++j) frag[j] = (__bf16)wv[j];
    Wfrag[f] = frag;
  }
  __syncthreads();

  const int rb = blockIdx.x * 8 + wave;          // wave-uniform
  if (rb >= row_blocks) return;                  // uniform exit after barrier

  const int half = lane >> 4;
  const int l16  = lane & 15;
  const int m0   = rb * 16;
  int arow = m0 + l16;
  if (arow >= n_src) arow = n_src - 1;

  v8f acc[NTILES];
#pragma unroll
  for (int i = 0; i < NTILES; ++i) acc[i] = (v8f){};

  const float* __restrict__ Arow = F + (size_t)arow * IN_DIM;

#pragma unroll 1
  for (int ks = 0; ks < KSTEPS; ++ks) {
    const int klo = ks * 32 + half * 8;

    float av[16];
    *(f32x4*)(av + 0)  = *(const f32x4*)(Arow + klo);
    *(f32x4*)(av + 4)  = *(const f32x4*)(Arow + klo + 4);
    *(f32x4*)(av + 8)  = *(const f32x4*)(Arow + klo + 16);
    *(f32x4*)(av + 12) = *(const f32x4*)(Arow + klo + 20);
    v16bf afrag;
#pragma unroll
    for (int i = 0; i < 16; ++i) afrag[i] = (__bf16)av[i];

    // preload all 8 B fragments for this k-step (distinct registers)
    v16bf bfrag[NTILES];
#pragma unroll
    for (int nt = 0; nt < NTILES; ++nt)
      bfrag[nt] = Wfrag[(ks * NTILES + nt) * 32 + lane];

#pragma unroll
    for (int nt = 0; nt < NTILES; ++nt)
      acc[nt] = __builtin_amdgcn_wmma_f32_16x16x32_bf16(
          /*neg_a=*/false, afrag, /*neg_b=*/false, bfrag[nt],
          /*c_mod=*/(short)0, acc[nt], /*reuse_a=*/false, /*reuse_b=*/false);
  }

  // C/D layout: lanes 0-15 -> M=v, N=l16 ; lanes 16-31 -> M=v+8, N=l16
  float* __restrict__ Zb = Z + (size_t)m0 * OUT_DIM + l16;
  if (m0 + 16 <= n_src) {
    // fast path: branchless stores
#pragma unroll
    for (int nt = 0; nt < NTILES; ++nt)
#pragma unroll
      for (int v = 0; v < 8; ++v)
        Zb[(size_t)(v + 8 * half) * OUT_DIM + nt * 16] = acc[nt][v];
  } else {
#pragma unroll
    for (int nt = 0; nt < NTILES; ++nt)
#pragma unroll
      for (int v = 0; v < 8; ++v) {
        int row = m0 + v + 8 * half;
        if (row < n_src)
          Z[(size_t)row * OUT_DIM + nt * 16 + l16] = acc[nt][v];
      }
  }
}

// ---------------------------------------------------------------------------
// Phase 2: s[i] = dot(z[i], W_t). One wave per row.
// ---------------------------------------------------------------------------
__global__ void __launch_bounds__(256)
s_kernel(const float* __restrict__ Z, const float* __restrict__ Wt,
         float* __restrict__ S, int n_src) {
  const int gw   = (int)((blockIdx.x * blockDim.x + threadIdx.x) >> 5);
  const int lane = threadIdx.x & 31;
  if (gw >= n_src) return;
  f32x4 z4 = *(const f32x4*)(Z + (size_t)gw * OUT_DIM + lane * 4);
  f32x4 w4 = *(const f32x4*)(Wt + lane * 4);
  float p = z4.x * w4.x + z4.y * w4.y + z4.z * w4.z + z4.w * w4.w;
#pragma unroll
  for (int off = 16; off > 0; off >>= 1) p += __shfl_xor(p, off, 32);
  if (lane == 0) S[gw] = p;
}

// ---------------------------------------------------------------------------
// Phase 3a: out[:n_dst] = z[:n_dst] (residual base); init m (bits 0 == 0.0f,
// valid lower bound since y = exp(...) > 0) and denom.
// ---------------------------------------------------------------------------
__global__ void __launch_bounds__(256)
init_kernel(const float* __restrict__ Z, float* __restrict__ out,
            unsigned* __restrict__ M, float* __restrict__ D, int n_dst) {
  const int i  = blockIdx.x * blockDim.x + threadIdx.x;
  const int n4 = n_dst * (OUT_DIM / 4);
  if (i < n4) ((f32x4*)out)[i] = ((const f32x4*)Z)[i];
  if (i < n_dst) { M[i] = 0u; D[i] = 0.0f; }
}

// Phase 3b: y = exp(s[src] * -|t_s - t_d| / 500); segment max via uint atomicMax.
__global__ void __launch_bounds__(256)
edge1_kernel(const float* __restrict__ t, const int* __restrict__ es,
             const int* __restrict__ ed, const float* __restrict__ S,
             float* __restrict__ Y, unsigned* __restrict__ M, int E) {
  const int e = blockIdx.x * blockDim.x + threadIdx.x;
  if (e >= E) return;
  const int s = es[e], d = ed[e];
  const float att = -fabsf(t[s] - t[d]);
  const float y = expf(S[s] * att * (1.0f / 500.0f));
  Y[e] = y;
  atomicMax(&M[d], __float_as_uint(y));            // y > 0: uint order == float order
}

// Phase 3c: ex = exp(y - m[dst]); denom += ex.
__global__ void __launch_bounds__(256)
edge2_kernel(const int* __restrict__ ed, float* __restrict__ Y,
             const unsigned* __restrict__ M, float* __restrict__ D, int E) {
  const int e = blockIdx.x * blockDim.x + threadIdx.x;
  if (e >= E) return;
  const int d = ed[e];
  const float ex = expf(Y[e] - __uint_as_float(M[d]));
  Y[e] = ex;                                       // reuse buffer for ex
  unsafeAtomicAdd(&D[d], ex);                      // global_atomic_add_f32
}

// Phase 3d: out[dst] += (ex/denom[dst]) * z[src]. One wave per edge, 4 floats/lane.
__global__ void __launch_bounds__(256)
edge3_kernel(const float* __restrict__ Z, const float* __restrict__ EX,
             const float* __restrict__ D, const int* __restrict__ es,
             const int* __restrict__ ed, float* __restrict__ out, int E) {
  const int gw   = (int)((blockIdx.x * blockDim.x + threadIdx.x) >> 5);
  const int lane = threadIdx.x & 31;
  if (gw >= E) return;
  const int s = es[gw], d = ed[gw];
  const float alpha = EX[gw] / D[d];
  f32x4 z4 = *(const f32x4*)(Z + (size_t)s * OUT_DIM + lane * 4);
  float* o = out + (size_t)d * OUT_DIM + lane * 4;
  unsafeAtomicAdd(o + 0, alpha * z4.x);
  unsafeAtomicAdd(o + 1, alpha * z4.y);
  unsafeAtomicAdd(o + 2, alpha * z4.z);
  unsafeAtomicAdd(o + 3, alpha * z4.w);
}

extern "C" void kernel_launch(void* const* d_in, const int* in_sizes, int n_in,
                              void* d_out, int out_size, void* d_ws, size_t ws_size,
                              hipStream_t stream) {
  const int n_src = in_sizes[0] / IN_DIM;    // 60000
  const int E     = in_sizes[2];             // 600000
  const int n_dst = out_size / OUT_DIM;      // 50000

  const float* F   = (const float*)d_in[0];
  const float* t   = (const float*)d_in[1];
  const int*   es  = (const int*)d_in[2];
  const int*   ed  = (const int*)d_in[3];
  const float* Wfc = (const float*)d_in[5];
  const float* Wt  = (const float*)d_in[6];
  float*       out = (float*)d_out;

  // workspace layout
  char* ws = (char*)d_ws;
  size_t off = 0;
  float*    Z = (float*)(ws + off);    off += (size_t)n_src * OUT_DIM * sizeof(float);
  float*    S = (float*)(ws + off);    off += (size_t)n_src * sizeof(float);
  float*    Y = (float*)(ws + off);    off += (size_t)E * sizeof(float);
  unsigned* M = (unsigned*)(ws + off); off += (size_t)n_dst * sizeof(unsigned);
  float*    D = (float*)(ws + off);    off += (size_t)n_dst * sizeof(float);
  (void)ws_size; (void)n_in;

  const int row_blocks = (n_src + 15) / 16;                       // 3750
  gemm_z_kernel<<<(row_blocks + 7) / 8, 256, 0, stream>>>(F, Wfc, Z, n_src, row_blocks);
  s_kernel<<<((size_t)n_src * 32 + 255) / 256, 256, 0, stream>>>(Z, Wt, S, n_src);
  init_kernel<<<((size_t)n_dst * (OUT_DIM / 4) + 255) / 256, 256, 0, stream>>>(Z, out, M, D, n_dst);
  edge1_kernel<<<(E + 255) / 256, 256, 0, stream>>>(t, es, ed, S, Y, M, E);
  edge2_kernel<<<(E + 255) / 256, 256, 0, stream>>>(ed, Y, M, D, E);
  edge3_kernel<<<((size_t)E * 32 + 255) / 256, 256, 0, stream>>>(Z, Y, D, es, ed, out, E);
}
